// SceneBoxEmb_17712445129342
// MI455X (gfx1250) — compile-verified, matrix operand order checked by reading, not run on hardware
//
#include <hip/hip_runtime.h>
#include <hip/hip_bf16.h>

typedef __attribute__((ext_vector_type(2)))  float    v2f;
typedef __attribute__((ext_vector_type(8)))  float    v8f;
typedef __attribute__((ext_vector_type(16))) _Float16 v16h;

#define UB     700
#define PN     256
#define SN     1024
#define CBOX   128
#define DSEED  256
#define CBFU   128
#define KDIM   512
#define ODIM   128
#define BPG    16          // boxes per workgroup
#define GPAD   8           // LDS row pad (floats) to break bank conflicts

__global__ __launch_bounds__(256)
void scenebox_fused_kernel(const float* __restrict__ union_box,
                           const float* __restrict__ box_features,
                           const float* __restrict__ agg_xyz,
                           const float* __restrict__ seed_feature,
                           const float* __restrict__ seed_xyz,
                           const float* __restrict__ bfu,
                           const float* __restrict__ W,
                           const float* __restrict__ bias,
                           float* __restrict__ out)
{
    __shared__ float        g[BPG][KDIM + GPAD];   // gfeat tile [16 x 512]
    __shared__ float        bmin[BPG][3], bmax[BPG][3];
    __shared__ unsigned int smask[SN];             // bit b: seed s inside box b
    __shared__ unsigned int pmask[PN];             // bit b: proposal p inside box b

    const int t    = threadIdx.x;
    const int box0 = blockIdx.x * BPG;
    const int lane = t & 31;
    const int wave = t >> 5;           // 8 waves -> 8 N-tiles of 16
    const int nbase = wave * 16;
    const int hsel  = lane >> 4;       // 0: lanes 0-15, 1: lanes 16-31
    const int l15   = lane & 15;

    // Warm L2/WGP$ for this wave's W rows while the max phase runs.
    __builtin_prefetch(&W[(nbase + l15) * KDIM], 0, 3);
    __builtin_prefetch(&W[(nbase + l15) * KDIM + 256], 0, 3);

    // ---- box AABBs (degenerate empty box for padded lanes) ----
    if (t < BPG) {
        const int b = box0 + t;
        float c0, c1, c2, s0, s1, s2;
        if (b < UB) {
            c0 = union_box[b*6+0]; c1 = union_box[b*6+1]; c2 = union_box[b*6+2];
            s0 = union_box[b*6+3]; s1 = union_box[b*6+4]; s2 = union_box[b*6+5];
        } else {
            c0 = c1 = c2 = 0.0f; s0 = s1 = s2 = -1.0f;   // min > max -> empty
        }
        bmin[t][0] = c0 - 0.5f*s0; bmin[t][1] = c1 - 0.5f*s1; bmin[t][2] = c2 - 0.5f*s2;
        bmax[t][0] = c0 + 0.5f*s0; bmax[t][1] = c1 + 0.5f*s1; bmax[t][2] = c2 + 0.5f*s2;
    }
    __syncthreads();

    // ---- inside masks: 16 box-bits per point ----
    for (int s = t; s < SN; s += 256) {
        const float x = seed_xyz[s*3+0], y = seed_xyz[s*3+1], z = seed_xyz[s*3+2];
        unsigned int m = 0;
#pragma unroll
        for (int b = 0; b < BPG; ++b) {
            const int in = (x >= bmin[b][0]) & (x <= bmax[b][0]) &
                           (y >= bmin[b][1]) & (y <= bmax[b][1]) &
                           (z >= bmin[b][2]) & (z <= bmax[b][2]);
            m |= (unsigned)in << b;
        }
        smask[s] = m;
    }
    {
        const int p = t;  // PN == 256 == blockDim.x
        const float x = agg_xyz[p*3+0], y = agg_xyz[p*3+1], z = agg_xyz[p*3+2];
        unsigned int m = 0;
#pragma unroll
        for (int b = 0; b < BPG; ++b) {
            const int in = (x >= bmin[b][0]) & (x <= bmax[b][0]) &
                           (y >= bmin[b][1]) & (y <= bmax[b][1]) &
                           (z >= bmin[b][2]) & (z <= bmax[b][2]);
            m |= (unsigned)in << b;
        }
        pmask[p] = m;
    }
    __syncthreads();

    // ---- g1: thread t handles seed-feature dim d = t for all 16 boxes ----
    // Exact semantics: max over s of (inside ? v : 0), init -inf.
    {
        const int d = t;
        float macc[BPG];
#pragma unroll
        for (int b = 0; b < BPG; ++b) macc[b] = -__builtin_inff();
        const float* __restrict__ row = seed_feature + (size_t)d * SN;  // contiguous in s
#pragma unroll 4
        for (int s = 0; s < SN; ++s) {
            const float v = row[s];
            const unsigned int m = smask[s];
#pragma unroll
            for (int b = 0; b < BPG; ++b) {
                const float vv = ((m >> b) & 1u) ? v : 0.0f;
                macc[b] = fmaxf(macc[b], vv);
            }
        }
#pragma unroll
        for (int b = 0; b < BPG; ++b) g[b][d] = macc[b];
    }

    // ---- g2: threads 0..127 handle box_features channel c = t ----
    if (t < CBOX) {
        const int c = t;
        float macc[BPG];
#pragma unroll
        for (int b = 0; b < BPG; ++b) macc[b] = -__builtin_inff();
#pragma unroll 4
        for (int p = 0; p < PN; ++p) {
            const float v = box_features[p * CBOX + c];   // coalesced across c
            const unsigned int m = pmask[p];
#pragma unroll
            for (int b = 0; b < BPG; ++b) {
                const float vv = ((m >> b) & 1u) ? v : 0.0f;
                macc[b] = fmaxf(macc[b], vv);
            }
        }
#pragma unroll
        for (int b = 0; b < BPG; ++b) g[b][DSEED + c] = macc[b];
    }

    // ---- box_feature_union copy into the tile ----
    for (int i = t; i < BPG * CBFU; i += 256) {
        const int b = i / CBFU, c = i % CBFU;
        const int gb = box0 + b;
        g[b][DSEED + CBOX + c] = (gb < UB) ? bfu[(size_t)gb * CBFU + c] : 0.0f;
    }
    __syncthreads();

    // ---- GEMM: out[16 x 128] = g[16 x 512] @ W[128 x 512]^T, one 16x16 tile/wave ----
    v8f acc = {};
#if defined(__AMDGCN__) && __has_builtin(__builtin_amdgcn_wmma_f32_16x16x4_f32)
    // Full-precision fp32 WMMA. A 16x4 frag: lanes 0-15 hold K=k0,k0+1;
    // lanes 16-31 hold K=k0+2,k0+3 (row M / col N = lane&15). Same for B.
    for (int k0 = 0; k0 < KDIM; k0 += 4) {
        const int ka = k0 + 2 * hsel;
        v2f a, b;
        a.x = g[l15][ka];
        a.y = g[l15][ka + 1];
        const float* __restrict__ wr = W + (size_t)(nbase + l15) * KDIM + ka;
        b.x = wr[0];
        b.y = wr[1];
        acc = __builtin_amdgcn_wmma_f32_16x16x4_f32(false, a, false, b,
                                                    (short)0, acc, false, false);
    }
#elif defined(__AMDGCN__)
    // Fallback: f16 WMMA (codegen-confirmed builtin), K-step 32.
    for (int k0 = 0; k0 < KDIM; k0 += 32) {
        v16h a, b;
#pragma unroll
        for (int v = 0; v < 8; ++v) {
#pragma unroll
            for (int h = 0; h < 2; ++h) {
                const int k = k0 + 2*v + h + 8 * (hsel + (v >= 4));
                a[2*v + h] = (_Float16)g[l15][k];
                b[2*v + h] = (_Float16)W[(size_t)(nbase + l15) * KDIM + k];
            }
        }
        acc = __builtin_amdgcn_wmma_f32_16x16x32_f16(false, a, false, b,
                                                     (short)0, acc, false, false);
    }
#endif

    // ---- epilogue: bias + sigmoid(log|x+1e-6|) == |x+1e-6|/(1+|x+1e-6|) ----
    const float bn = bias[nbase + l15];
#pragma unroll
    for (int j = 0; j < 8; ++j) {
        const int m  = j + 8 * hsel;       // C/D layout: VGPR j -> row j (+8 for hi lanes)
        const int gm = box0 + m;
        if (gm < UB) {
            const float x  = acc[j] + bn + 1e-6f;
            const float ax = fabsf(x);
            out[(size_t)gm * ODIM + nbase + l15] = ax / (1.0f + ax);
        }
    }
}

extern "C" void kernel_launch(void* const* d_in, const int* in_sizes, int n_in,
                              void* d_out, int out_size, void* d_ws, size_t ws_size,
                              hipStream_t stream) {
    const float* union_box    = (const float*)d_in[0];
    const float* box_features = (const float*)d_in[1];
    const float* agg_xyz      = (const float*)d_in[2];
    const float* seed_feature = (const float*)d_in[3];
    const float* seed_xyz     = (const float*)d_in[4];
    const float* bfu          = (const float*)d_in[5];
    const float* W_gs         = (const float*)d_in[6];
    const float* b_gs         = (const float*)d_in[7];
    float* out = (float*)d_out;

    const int grid = (UB + BPG - 1) / BPG;   // 44 workgroups, 8 waves each
    scenebox_fused_kernel<<<grid, 256, 0, stream>>>(
        union_box, box_features, agg_xyz, seed_feature, seed_xyz,
        bfu, W_gs, b_gs, out);
}
